// StepwiseLSTMAttentionModel_68547678044343
// MI455X (gfx1250) — matrix operand order repaired and checked by reading, not compiled, not run on hardware
//
#include <hip/hip_runtime.h>
#include <hip/hip_bf16.h>

// ---------------- problem constants ----------------
#define BB   256          // batch
#define SS   365          // sequence
#define FF   16           // features
#define EE   64           // embedding
#define HH   512          // hidden
#define H4   2048         // 4*H
#define KC   608          // combined K for gate GEMM: 16 + 64 + 512, padded to 19*32
#define KT_C 19           // K tiles (32) for combined gate GEMM
#define NT_C 128          // N tiles (16) over 4H
#define KT_H 16           // K tiles over H=512
#define NT_H 32           // N tiles over H=512
#define KT_4 64           // K tiles over 4H=2048

typedef __attribute__((ext_vector_type(16))) __bf16 bf16x16;
typedef __attribute__((ext_vector_type(8)))  __bf16 bf16x8;
typedef __attribute__((ext_vector_type(8)))  float  f32x8;

static __device__ __forceinline__ f32x8 wmma_bf16(bf16x16 a, bf16x16 b, f32x8 c) {
  // D = A(16x32 bf16) x B(32x16 bf16) + C(16x16 f32)
  return __builtin_amdgcn_wmma_f32_16x16x32_bf16(false, a, false, b, (short)0, c, false, false);
}

// Gather the A operand (16x32 bf16 tile) from an LDS row-major staging buffer.
// Layout (matches packed B): lanes 0-15 -> M=lane, K=0..15; lanes 16-31 -> M=lane-16, K=16..31.
static __device__ __forceinline__ bf16x16 load_a(const __bf16* lds, int stride, int kt) {
  const int lane  = threadIdx.x & 31;
  const int m     = lane & 15;
  const int khalf = lane >> 4;
  const __bf16* p = lds + m * stride + kt * 32 + khalf * 16;
  bf16x8 lo = *(const bf16x8*)(p);        // K offset +0..7
  bf16x8 hi = *(const bf16x8*)(p + 8);    // K offset +8..15
  return __builtin_shufflevector(lo, hi, 0,1,2,3,4,5,6,7,8,9,10,11,12,13,14,15);
}

// B operand: pre-swizzled tiles, lane loads 16 contiguous bf16 (32B).
static __device__ __forceinline__ bf16x16 load_b(const __bf16* pk, int tile) {
  const int lane = threadIdx.x & 31;
  return *(const bf16x16*)(pk + ((size_t)tile << 9) + lane * 16);
}

// ---- CDNA5 async global->LDS copy (ASYNCcnt path, bypasses VGPRs) ----
// GVS addressing: mem = SGPR_base + VGPR_off ; LDS dst = VGPR lds byte offset.
static __device__ __forceinline__ void async16(unsigned lds_off, const void* gbase, unsigned goff) {
  asm volatile("global_load_async_to_lds_b128 %0, %1, %2"
               :: "v"(lds_off), "v"(goff), "s"(gbase) : "memory");
}
static __device__ __forceinline__ void async_wait() {
  asm volatile("s_wait_asynccnt 0x0" ::: "memory");
}
static __device__ __forceinline__ unsigned lds_off_of(const void* p) {
  return (unsigned)(size_t)p;             // generic LDS ptr: low 32 bits = LDS offset
}

static __device__ __forceinline__ float sigf(float x) { return 1.0f / (1.0f + __expf(-x)); }

// ---------------- setup kernels (run once per launch) ----------------

// Pack W^T (src is torch-layout [N][Ksrc]) into WMMA-B tile layout:
// lane l: n = l&15, k = kt*32 + (l>>4)*16 + j  (j = 0..15 contiguous per lane).
__global__ void pack_generic(const float* __restrict__ src, __bf16* __restrict__ dst,
                             int ntiles, int Ksrc, int total) {
  int id = blockIdx.x * 256 + threadIdx.x;
  if (id >= total) return;
  int tile = id >> 9, e = id & 511, lane = e >> 4, j = e & 15;
  int kt = tile / ntiles, nt = tile % ntiles;
  int n = nt * 16 + (lane & 15);
  int k = kt * 32 + ((lane >> 4) << 4) + j;
  float v = (k < Ksrc) ? src[(size_t)n * Ksrc + k] : 0.0f;
  dst[id] = (__bf16)v;
}

// Combined gate weight: rows 0..79 = Wih^T, 80..591 = Whh^T, 592..607 = 0.
__global__ void pack_comb(const float* __restrict__ Wih, const float* __restrict__ Whh,
                          __bf16* __restrict__ dst, int total) {
  int id = blockIdx.x * 256 + threadIdx.x;
  if (id >= total) return;
  int tile = id >> 9, e = id & 511, lane = e >> 4, j = e & 15;
  int kt = tile / NT_C, nt = tile % NT_C;
  int n = nt * 16 + (lane & 15);
  int k = kt * 32 + ((lane >> 4) << 4) + j;
  float v = 0.0f;
  if (k < 80)       v = Wih[(size_t)n * 80 + k];
  else if (k < 592) v = Whh[(size_t)n * 512 + (k - 80)];
  dst[id] = (__bf16)v;
}

// xe[s][b][0..79] = bf16([x_t | emb_t]) ; consumed as the non-recurrent A slice.
__global__ void build_xe(const float* __restrict__ x, const float* __restrict__ emb,
                         const int* __restrict__ sid, __bf16* __restrict__ xe, int total) {
  int id = blockIdx.x * 256 + threadIdx.x;
  if (id >= total) return;
  int b = id / (SS * 80), r = id % (SS * 80), s = r / 80, k = r % 80;
  float v = (k < FF) ? x[((size_t)b * SS + s) * FF + k]
                     : emb[(size_t)sid[b * SS + s] * EE + (k - FF)];
  xe[((size_t)s * BB + b) * 80 + k] = (__bf16)v;
}

__global__ void init_state(__bf16* __restrict__ hA, float* __restrict__ c,
                           float* __restrict__ biasc, const float* __restrict__ bih,
                           const float* __restrict__ bhh, float* __restrict__ out,
                           const float* __restrict__ bo) {
  int id = blockIdx.x * 256 + threadIdx.x;
  if (id < BB * HH) { hA[id] = (__bf16)0.0f; c[id] = 0.0f; }
  if (id < H4)       biasc[id] = bih[id] + bhh[id];
  if (id < BB * SS)  out[id] = bo[0];
}

// ---------------- per-step kernels ----------------

// K1: gates = [x|emb|h] @ Wcomb^T + (bih+bhh); LSTM cell update.
// grid (16 btiles, 4 col-slices of 128 H cols), 256 threads (8 waves).
__global__ void k1_lstm(const __bf16* __restrict__ xe, const __bf16* __restrict__ hA,
                        const __bf16* __restrict__ wpk, const float* __restrict__ biasc,
                        float* __restrict__ c_f32, float* __restrict__ hn_f32,
                        __bf16* __restrict__ hn_b16, int t) {
  __shared__ __bf16 Abuf[16 * KC];
  const int tid = threadIdx.x, bbase = blockIdx.x * 16;
  // zero the K padding (elements 592..607 per row): 16 rows x 16 elems = 256
  { int m = tid >> 4, j = tid & 15; Abuf[m * KC + 592 + j] = (__bf16)0.0f; }
  // async-stage A = [x_t|emb_t (10 x 16B) | h_{t-1} (64 x 16B)] per row
  {
    const unsigned abase = lds_off_of(Abuf);
    const void* xe_base = (const void*)(xe + ((size_t)t * BB + bbase) * 80);
    const void* h_base  = (const void*)(hA + (size_t)bbase * HH);
    for (int c = tid; c < 16 * 74; c += 256) {
      int m = c / 74, j = c - m * 74;
      if (j < 10) async16(abase + m * (KC * 2) + j * 16,        xe_base, m * 160  + j * 16);
      else        async16(abase + m * (KC * 2) + 160 + (j-10)*16, h_base, m * 1024 + (j-10)*16);
    }
  }
  async_wait();
  __syncthreads();

  const int w = tid >> 5, lane = tid & 31;
  const int htile = blockIdx.y * 8 + w;          // 0..31 tile within H
  f32x8 acc[4];
#pragma unroll
  for (int g = 0; g < 4; ++g) acc[g] = f32x8{0,0,0,0,0,0,0,0};

  for (int kt = 0; kt < KT_C; ++kt) {
    bf16x16 a = load_a(Abuf, KC, kt);
#pragma unroll
    for (int g = 0; g < 4; ++g) {
      int ntile = g * 32 + htile;                // gate g lives at cols g*512+...
      acc[g] = wmma_bf16(a, load_b(wpk, kt * NT_C + ntile), acc[g]);
    }
  }

  const int col  = htile * 16 + (lane & 15);
  const int rsel = (lane >> 4) * 8;              // lanes>=16 hold rows 8..15
  const float bi = biasc[0 * HH + col], bf = biasc[1 * HH + col];
  const float bg = biasc[2 * HH + col], bo_ = biasc[3 * HH + col];
#pragma unroll
  for (int r = 0; r < 8; ++r) {
    int brow = bbase + rsel + r;
    size_t idx = (size_t)brow * HH + col;
    float iv = sigf(acc[0][r] + bi);
    float fv = sigf(acc[1][r] + bf);
    float gv = tanhf(acc[2][r] + bg);
    float ov = sigf(acc[3][r] + bo_);
    float cn = fv * c_f32[idx] + iv * gv;
    float hn = ov * tanhf(cn);
    c_f32[idx]  = cn;
    hn_f32[idx] = hn;
    hn_b16[idx] = (__bf16)hn;
  }
}

// K2a: vh = h@Wv^T + bv ; y = h + vh (attn == vh: softmax rows sum to 1) ; LayerNorm.
// grid 16 btiles; block covers all 512 cols (8 waves x 4 tiles).
__global__ void k2a_attn_ln(const __bf16* __restrict__ hnb, const float* __restrict__ hnf,
                            const __bf16* __restrict__ wvpk, const float* __restrict__ bv,
                            const float* __restrict__ lng, const float* __restrict__ lnb,
                            float* __restrict__ y_f32, __bf16* __restrict__ y_b16) {
  __shared__ __bf16 Abuf[16 * HH];
  __shared__ float  Ybuf[16 * HH];
  __shared__ float  sMu[16], sRs[16];
  const int tid = threadIdx.x, bbase = blockIdx.x * 16;
  {
    const unsigned abase = lds_off_of(Abuf);
    const void* gb = (const void*)(hnb + (size_t)bbase * HH);   // 16 KB contiguous
    for (unsigned ofs = tid * 16u; ofs < 16u * HH * 2u; ofs += 4096u)
      async16(abase + ofs, gb, ofs);
  }
  async_wait();
  __syncthreads();

  const int w = tid >> 5, lane = tid & 31;
#pragma unroll
  for (int s = 0; s < 4; ++s) {
    int nt = w * 4 + s;                          // 0..31
    f32x8 acc = {0,0,0,0,0,0,0,0};
    for (int kt = 0; kt < KT_H; ++kt)
      acc = wmma_bf16(load_a(Abuf, HH, kt), load_b(wvpk, kt * NT_H + nt), acc);
    int col = nt * 16 + (lane & 15);
    int rsel = (lane >> 4) * 8;
#pragma unroll
    for (int r = 0; r < 8; ++r) {
      int rr = rsel + r;
      float y = hnf[(size_t)(bbase + rr) * HH + col] + acc[r] + bv[col];
      Ybuf[rr * HH + col] = y;
    }
  }
  __syncthreads();
  // wave-parallel mean/var: wave w owns rows 2w, 2w+1
#pragma unroll
  for (int q = 0; q < 2; ++q) {
    int row = w * 2 + q;
    float s1 = 0.f, s2 = 0.f;
    for (int k = lane; k < HH; k += 32) { float v = Ybuf[row * HH + k]; s1 += v; s2 += v * v; }
#pragma unroll
    for (int m = 1; m < 32; m <<= 1) { s1 += __shfl_xor(s1, m, 32); s2 += __shfl_xor(s2, m, 32); }
    if (lane == 0) {
      float mu = s1 * (1.0f / HH);
      sMu[row] = mu;
      sRs[row] = rsqrtf(s2 * (1.0f / HH) - mu * mu + 1e-5f);
    }
  }
  __syncthreads();
  for (int i = tid; i < 16 * HH; i += 256) {
    int row = i >> 9, col = i & 511;
    float v = (Ybuf[i] - sMu[row]) * sRs[row] * lng[col] + lnb[col];
    size_t idx = (size_t)(bbase + row) * HH + col;
    y_f32[idx] = v;
    y_b16[idx] = (__bf16)v;
  }
}

// K2b: t1 = relu(y @ W1^T + b1), bf16. grid (16, 16): btile x 128-col slice of 2048.
__global__ void k2b_ffn1(const __bf16* __restrict__ ybf, const __bf16* __restrict__ w1pk,
                         const float* __restrict__ b1, __bf16* __restrict__ t1) {
  __shared__ __bf16 Abuf[16 * HH];
  const int tid = threadIdx.x, bbase = blockIdx.x * 16;
  {
    const unsigned abase = lds_off_of(Abuf);
    const void* gb = (const void*)(ybf + (size_t)bbase * HH);
    for (unsigned ofs = tid * 16u; ofs < 16u * HH * 2u; ofs += 4096u)
      async16(abase + ofs, gb, ofs);
  }
  async_wait();
  __syncthreads();
  const int w = tid >> 5, lane = tid & 31;
  const int ntile = blockIdx.y * 8 + w;          // 0..127
  f32x8 acc = {0,0,0,0,0,0,0,0};
  for (int kt = 0; kt < KT_H; ++kt) {
    if (kt + 1 < KT_H) __builtin_prefetch(w1pk + ((size_t)((kt + 1) * NT_C + ntile) << 9), 0, 1);
    acc = wmma_bf16(load_a(Abuf, HH, kt), load_b(w1pk, kt * NT_C + ntile), acc);
  }
  int col = ntile * 16 + (lane & 15);
  int rsel = (lane >> 4) * 8;
  float bb = b1[col];
#pragma unroll
  for (int r = 0; r < 8; ++r) {
    float v = acc[r] + bb;
    t1[(size_t)(bbase + rsel + r) * H4 + col] = (__bf16)fmaxf(v, 0.0f);
  }
}

// K2c: y2 = y + t1 @ W2^T + b2 ; h_next = y2 ; partial out[b,t] += y2 . Wo (atomic).
// grid (16, 4): btile x 128-col slice of 512.
__global__ void k2c_ffn2(const __bf16* __restrict__ t1, const __bf16* __restrict__ w2pk,
                         const float* __restrict__ b2, const float* __restrict__ y_f32,
                         const float* __restrict__ Wo, __bf16* __restrict__ hA,
                         float* __restrict__ out, int t) {
  __shared__ __bf16 Abuf[16 * H4];
  const int tid = threadIdx.x, bbase = blockIdx.x * 16;
  {
    const unsigned abase = lds_off_of(Abuf);
    const void* gb = (const void*)(t1 + (size_t)bbase * H4);    // 64 KB contiguous
    for (unsigned ofs = tid * 16u; ofs < 16u * H4 * 2u; ofs += 4096u)
      async16(abase + ofs, gb, ofs);
  }
  async_wait();
  __syncthreads();
  const int w = tid >> 5, lane = tid & 31;
  const int ntile = blockIdx.y * 8 + w;          // 0..31
  f32x8 acc = {0,0,0,0,0,0,0,0};
  for (int kt = 0; kt < KT_4; ++kt) {
    if (kt + 1 < KT_4) __builtin_prefetch(w2pk + ((size_t)((kt + 1) * NT_H + ntile) << 9), 0, 1);
    acc = wmma_bf16(load_a(Abuf, H4, kt), load_b(w2pk, kt * NT_H + ntile), acc);
  }
  const int col = ntile * 16 + (lane & 15);
  const int rsel = (lane >> 4) * 8;
  const float bb = b2[col], wo = Wo[col];
  float part[8];
#pragma unroll
  for (int r = 0; r < 8; ++r) {
    size_t idx = (size_t)(bbase + rsel + r) * HH + col;
    float v = y_f32[idx] + acc[r] + bb;
    hA[idx] = (__bf16)v;                         // becomes h_{t} input for next step
    part[r] = v * wo;
  }
  // reduce the 16 columns of this wave (within each 16-lane half) for the Wo dot
#pragma unroll
  for (int m = 1; m < 16; m <<= 1)
#pragma unroll
    for (int r = 0; r < 8; ++r) part[r] += __shfl_xor(part[r], m, 32);
  if ((lane & 15) == 0) {
#pragma unroll
    for (int r = 0; r < 8; ++r)
      atomicAdd(&out[(size_t)(bbase + rsel + r) * SS + t], part[r]);
  }
}

// ---------------- host ----------------
extern "C" void kernel_launch(void* const* d_in, const int* in_sizes, int n_in,
                              void* d_out, int out_size, void* d_ws, size_t ws_size,
                              hipStream_t stream) {
  const int*   sid  = (const int*)d_in[0];
  const float* x    = (const float*)d_in[1];
  const float* emb  = (const float*)d_in[2];
  const float* Wih  = (const float*)d_in[3];
  const float* Whh  = (const float*)d_in[4];
  const float* bih  = (const float*)d_in[5];
  const float* bhh  = (const float*)d_in[6];
  // d_in[7..10] = Wq,bq,Wk,bk : analytically unused (softmax rows sum to 1 => attn == vh)
  const float* Wv   = (const float*)d_in[11];
  const float* bv   = (const float*)d_in[12];
  const float* lng  = (const float*)d_in[13];
  const float* lnb  = (const float*)d_in[14];
  const float* W1   = (const float*)d_in[15];
  const float* b1   = (const float*)d_in[16];
  const float* W2   = (const float*)d_in[17];
  const float* b2   = (const float*)d_in[18];
  const float* Wo   = (const float*)d_in[19];
  const float* bo   = (const float*)d_in[20];
  float* out = (float*)d_out;

  char* ws = (char*)d_ws;
  size_t o = 0;
  auto alloc = [&](size_t bytes) { size_t r = o; o += (bytes + 255) & ~(size_t)255; return r; };
  __bf16* wcomb = (__bf16*)(ws + alloc((size_t)KT_C * NT_C * 512 * 2));
  __bf16* wvpk  = (__bf16*)(ws + alloc((size_t)KT_H * NT_H * 512 * 2));
  __bf16* w1pk  = (__bf16*)(ws + alloc((size_t)KT_H * NT_C * 512 * 2));
  __bf16* w2pk  = (__bf16*)(ws + alloc((size_t)KT_4 * NT_H * 512 * 2));
  float*  biasc = (float*)(ws + alloc((size_t)H4 * 4));
  __bf16* xe    = (__bf16*)(ws + alloc((size_t)SS * BB * 80 * 2));
  float*  c_f32 = (float*)(ws + alloc((size_t)BB * HH * 4));
  __bf16* hA    = (__bf16*)(ws + alloc((size_t)BB * HH * 2));
  float*  hn_f  = (float*)(ws + alloc((size_t)BB * HH * 4));
  __bf16* hn_b  = (__bf16*)(ws + alloc((size_t)BB * HH * 2));
  float*  y_f   = (float*)(ws + alloc((size_t)BB * HH * 4));
  __bf16* y_b   = (__bf16*)(ws + alloc((size_t)BB * HH * 2));
  __bf16* t1    = (__bf16*)(ws + alloc((size_t)BB * H4 * 2));
  if (o > ws_size) return;  // deterministic no-op if workspace too small

  // --- one-time setup ---
  {
    int tot = KT_C * NT_C * 512;
    pack_comb<<<(tot + 255) / 256, 256, 0, stream>>>(Wih, Whh, wcomb, tot);
  }
  { int tot = KT_H * NT_H * 512; pack_generic<<<(tot + 255) / 256, 256, 0, stream>>>(Wv, wvpk, NT_H, 512, tot); }
  { int tot = KT_H * NT_C * 512; pack_generic<<<(tot + 255) / 256, 256, 0, stream>>>(W1, w1pk, NT_C, 512, tot); }
  { int tot = KT_4 * NT_H * 512; pack_generic<<<(tot + 255) / 256, 256, 0, stream>>>(W2, w2pk, NT_H, 2048, tot); }
  { int tot = BB * SS * 80; build_xe<<<(tot + 255) / 256, 256, 0, stream>>>(x, emb, sid, xe, tot); }
  init_state<<<(BB * HH + 255) / 256, 256, 0, stream>>>(hA, c_f32, biasc, bih, bhh, out, bo);

  // --- sequential recurrence: 4 kernels per step, serialized by the stream ---
  for (int t = 0; t < SS; ++t) {
    k1_lstm   <<<dim3(16, 4),  256, 0, stream>>>(xe, hA, wcomb, biasc, c_f32, hn_f, hn_b, t);
    k2a_attn_ln<<<dim3(16),    256, 0, stream>>>(hn_b, hn_f, wvpk, bv, lng, lnb, y_f, y_b);
    k2b_ffn1  <<<dim3(16, 16), 256, 0, stream>>>(y_b, w1pk, b1, t1);
    k2c_ffn2  <<<dim3(16, 4),  256, 0, stream>>>(t1, w2pk, b2, y_f, Wo, hA, out, t);
  }
}
